// ProteinGN_29635274342505
// MI455X (gfx1250) — compile-verified, matrix operand order checked by reading, not run on hardware
//
#include <hip/hip_runtime.h>
#include <hip/hip_bf16.h>

typedef __attribute__((ext_vector_type(16))) _Float16 v16h;
typedef __attribute__((ext_vector_type(8)))  _Float16 v8h;
typedef __attribute__((ext_vector_type(4)))  _Float16 v4h;
typedef __attribute__((ext_vector_type(8)))  float    v8f;

#define N_NODES 102400
#define N_GRAPH 64
#define NPG     1600
#define N_EDGES 1638400
#define EPG     25600

// K index of slot h (0..15) of a v16h WMMA operand (CDNA5 16-bit operand layout):
// slots 0..7  -> K =  h      + 8*laneHi
// slots 8..15 -> K = 16+(h-8)+ 8*laneHi   (two contiguous 8-half runs per lane)
__device__ __forceinline__ int kslot(int h, int laneHi) {
    return (h < 8) ? (h + 8 * laneHi) : (16 + (h - 8) + 8 * laneHi);
}

__device__ __forceinline__ float sigmoidf_(float x) { return 1.0f / (1.0f + __expf(-x)); }

// Load one lane's A fragment for K32 chunk `ch` from an LDS row (two ds_load_b128).
__device__ __forceinline__ v16h load_a_frag(const _Float16* rowp, int ch, int laneHi) {
    const v8h lo = *(const v8h*)(rowp + 32 * ch + 8 * laneHi);
    const v8h hi = *(const v8h*)(rowp + 32 * ch + 16 + 8 * laneHi);
    v16h a;
#pragma unroll
    for (int i = 0; i < 8; ++i) { a[i] = lo[i]; a[8 + i] = hi[i]; }
    return a;
}

__device__ __forceinline__ v4h cvt4(const float4 f) {
    v4h h;
    h[0] = (_Float16)f.x; h[1] = (_Float16)f.y;
    h[2] = (_Float16)f.z; h[3] = (_Float16)f.w;
    return h;
}

// ---------------- K0: pack all B-matrix WMMA fragments into per-lane layout ----------------
// frag 0,1 : encoder edge  [Wes | Wee@k=60], K=64 (2 chunks), 8 out cols
// frag 2..4: encoder node  [Wnn | Wni],      K=96 (3 chunks), 16 out cols
// frag 5   : hidden edge   [hWee | hWes],    K=32, 8 out cols
// frag 6   : hidden node   [hWnn | hWni],    K=32, 16 out cols
__global__ __launch_bounds__(256) void k_pack_b(
    const float* __restrict__ Wee, const float* __restrict__ Wes,
    const float* __restrict__ Wnn, const float* __restrict__ Wni,
    const float* __restrict__ hWee, const float* __restrict__ hWes,
    const float* __restrict__ hWnn, const float* __restrict__ hWni,
    _Float16* __restrict__ Bp)
{
    const int tid = threadIdx.x;
    if (tid >= 7 * 32) return;
    const int f = tid >> 5, lane = tid & 31;
    const int laneHi = lane >> 4, col = lane & 15;
    for (int h = 0; h < 16; ++h) {
        const int ks = kslot(h, laneHi);
        float v = 0.f;
        if (f < 2) {
            int k = 32 * f + ks;
            if (col < 8) {
                if (k < 60)       v = Wes[col * 60 + k];
                else if (k == 60) v = Wee[col];
            }
        } else if (f < 5) {
            int k = 32 * (f - 2) + ks;
            if (k < 60)      v = Wnn[col * 60 + k];
            else if (k < 68) v = Wni[col * 8 + (k - 60)];
        } else if (f == 5) {
            int k = ks;
            if (col < 8) {
                if (k < 8)       v = hWee[col * 8 + k];
                else if (k < 24) v = hWes[col * 16 + (k - 8)];
            }
        } else {
            int k = ks;
            if (k < 16)      v = hWnn[col * 16 + k];
            else if (k < 24) v = hWni[col * 8 + (k - 16)];
        }
        Bp[(f * 32 + lane) * 16 + h] = (_Float16)v;
    }
}

// ---------------- K1: edge encoder  e = relu([nf[senders] | ef] @ B + be) ----------------
__global__ __launch_bounds__(256) void k_edge_enc(
    const float* __restrict__ nf, const float* __restrict__ ef,
    const int* __restrict__ senders,
    const float* __restrict__ be, const _Float16* __restrict__ Bp,
    _Float16* __restrict__ e1)
{
    __shared__ __align__(16) _Float16 sA[128][64];
    __shared__ int sSend[128];
    const int tid = threadIdx.x;
    const long e0 = (long)blockIdx.x * 128;
    if (tid < 128) sSend[tid] = senders[e0 + tid];
    __syncthreads();
    // 15 float4 loads per gathered row (nf rows are 240B = 16B-aligned), 8B LDS stores
    for (int idx = tid; idx < 128 * 15; idx += 256) {
        int row = idx / 15, q = idx - row * 15;
        const float4 f = *(const float4*)&nf[(long)sSend[row] * 60 + 4 * q];
        *(v4h*)&sA[row][4 * q] = cvt4(f);
    }
    if (tid < 128) {
        v4h t = {};
        t[0] = (_Float16)ef[e0 + tid];          // ef at k=60, k=61..63 zero pad
        *(v4h*)&sA[tid][60] = t;
    }
    __syncthreads();
    const int wave = tid >> 5, lane = tid & 31;
    const int laneHi = lane >> 4, col = lane & 15;
    const _Float16* rowp = &sA[wave * 16 + col][0];
    const v16h* B = (const v16h*)Bp;
    v8f c = {};
#pragma unroll
    for (int ch = 0; ch < 2; ++ch) {
        v16h a = load_a_frag(rowp, ch, laneHi);
        v16h b = B[ch * 32 + lane];
        c = __builtin_amdgcn_wmma_f32_16x16x32_f16(false, a, false, b, (short)0, c, false, false);
    }
    const float bias = be[col & 7];
    __syncthreads();
#pragma unroll
    for (int r = 0; r < 8; ++r)
        sA[wave * 16 + r + 8 * laneHi][col] = (_Float16)fmaxf(c[r] + bias, 0.f);
    __syncthreads();
    if (tid < 128) *(v8h*)&e1[(e0 + tid) * 8] = *(const v8h*)&sA[tid][0];
}

// ---------------- K2: per-graph segment-mean of edge latents into receiver nodes ----------------
__global__ __launch_bounds__(256) void k_inc_mean(
    const _Float16* __restrict__ e, const int* __restrict__ receivers,
    float* __restrict__ inc)
{
    __shared__ float sSum[NPG * 8];   // 51.2 KB
    __shared__ float sCnt[NPG];       //  6.4 KB
    const int g = blockIdx.x, tid = threadIdx.x;
    for (int i = tid; i < NPG * 8; i += 256) sSum[i] = 0.f;
    for (int i = tid; i < NPG; i += 256) sCnt[i] = 0.f;
    __syncthreads();
    const long ebase = (long)g * EPG;
    const int nbase = g * NPG;
    for (int i = tid; i < EPG; i += 256) {
        const long eid = ebase + i;
        const int r = receivers[eid] - nbase;
        const v8h ev = *(const v8h*)&e[eid * 8];
#pragma unroll
        for (int j = 0; j < 8; ++j)
            atomicAdd(&sSum[r * 8 + j], (float)ev[j]);
        atomicAdd(&sCnt[r], 1.0f);
    }
    __syncthreads();
    for (int i = tid; i < NPG * 2; i += 256) {
        int node = i >> 1, q = i & 1;
        float den = fmaxf(sCnt[node], 1.0f);
        float4 o;
        o.x = sSum[node * 8 + 4 * q + 0] / den;
        o.y = sSum[node * 8 + 4 * q + 1] / den;
        o.z = sSum[node * 8 + 4 * q + 2] / den;
        o.w = sSum[node * 8 + 4 * q + 3] / den;
        *(float4*)&inc[(long)(nbase + node) * 8 + 4 * q] = o;
    }
}

// ---------------- K3: node encoder  n = relu([nf | inc] @ B + bn) ----------------
__global__ __launch_bounds__(256) void k_node_enc(
    const float* __restrict__ nf, const float* __restrict__ inc,
    const float* __restrict__ bn, const _Float16* __restrict__ Bp,
    _Float16* __restrict__ n1)
{
    __shared__ __align__(16) _Float16 sA[128][96];
    const int tid = threadIdx.x;
    const long n0 = (long)blockIdx.x * 128;
    for (int idx = tid; idx < 128 * 15; idx += 256) {
        int row = idx / 15, q = idx - row * 15;
        const float4 f = *(const float4*)&nf[(n0 + row) * 60 + 4 * q];
        *(v4h*)&sA[row][4 * q] = cvt4(f);
    }
    for (int idx = tid; idx < 128 * 2; idx += 256) {
        int row = idx >> 1, q = idx & 1;
        const float4 f = *(const float4*)&inc[(n0 + row) * 8 + 4 * q];
        *(v4h*)&sA[row][60 + 4 * q] = cvt4(f);
    }
    for (int idx = tid; idx < 128 * 7; idx += 256) {
        int row = idx / 7, q = idx - row * 7;
        v4h z = {};
        *(v4h*)&sA[row][68 + 4 * q] = z;
    }
    __syncthreads();
    const int wave = tid >> 5, lane = tid & 31;
    const int laneHi = lane >> 4, col = lane & 15;
    const _Float16* rowp = &sA[wave * 16 + col][0];
    const v16h* B = (const v16h*)Bp;
    v8f c = {};
#pragma unroll
    for (int ch = 0; ch < 3; ++ch) {
        v16h a = load_a_frag(rowp, ch, laneHi);
        v16h b = B[(2 + ch) * 32 + lane];
        c = __builtin_amdgcn_wmma_f32_16x16x32_f16(false, a, false, b, (short)0, c, false, false);
    }
    const float bias = bn[col];
    __syncthreads();
#pragma unroll
    for (int r = 0; r < 8; ++r)
        sA[wave * 16 + r + 8 * laneHi][col] = (_Float16)fmaxf(c[r] + bias, 0.f);
    __syncthreads();
    {
        int row = tid >> 1, half = tid & 1;
        *(v8h*)&n1[(n0 + row) * 16 + 8 * half] = *(const v8h*)&sA[row][8 * half];
    }
}

// ---------------- K4: global encoder + fold g into per-graph hidden biases ----------------
__global__ __launch_bounds__(256) void k_glob_enc(
    const _Float16* __restrict__ e1, const _Float16* __restrict__ n1,
    const float* __restrict__ Wge, const float* __restrict__ Wgn, const float* __restrict__ bg,
    const float* __restrict__ Weg, const float* __restrict__ be_h,
    const float* __restrict__ Wng, const float* __restrict__ bn_h,
    float* __restrict__ g1, float* __restrict__ gb_e, float* __restrict__ gb_n)
{
    __shared__ float acc[24];
    const int g = blockIdx.x, tid = threadIdx.x;
    if (tid < 24) acc[tid] = 0.f;
    __syncthreads();
    float es[8]; for (int j = 0; j < 8; ++j) es[j] = 0.f;
    const long ebase = (long)g * EPG;
    for (int i = tid; i < EPG; i += 256) {
        const v8h ev = *(const v8h*)&e1[(ebase + i) * 8];
#pragma unroll
        for (int j = 0; j < 8; ++j) es[j] += (float)ev[j];
    }
    float ns[16]; for (int j = 0; j < 16; ++j) ns[j] = 0.f;
    const long nbase = (long)g * NPG;
    for (int i = tid; i < NPG; i += 256) {
        const v8h a = *(const v8h*)&n1[(nbase + i) * 16];
        const v8h b = *(const v8h*)&n1[(nbase + i) * 16 + 8];
#pragma unroll
        for (int j = 0; j < 8; ++j) { ns[j] += (float)a[j]; ns[8 + j] += (float)b[j]; }
    }
#pragma unroll
    for (int j = 0; j < 8; ++j) atomicAdd(&acc[j], es[j]);
#pragma unroll
    for (int j = 0; j < 16; ++j) atomicAdd(&acc[8 + j], ns[j]);
    __syncthreads();
    if (tid == 0) {
        float gv[4];
        for (int o = 0; o < 4; ++o) {
            float s = bg[o];
            for (int j = 0; j < 8;  ++j) s += Wge[o * 8 + j]  * (acc[j] / (float)EPG);
            for (int j = 0; j < 16; ++j) s += Wgn[o * 16 + j] * (acc[8 + j] / (float)NPG);
            gv[o] = fmaxf(s, 0.f);
            g1[g * 4 + o] = gv[o];
        }
        for (int o = 0; o < 8; ++o) {
            float s = be_h[o];
            for (int k = 0; k < 4; ++k) s += Weg[o * 4 + k] * gv[k];
            gb_e[g * 8 + o] = s;
        }
        for (int o = 0; o < 16; ++o) {
            float s = bn_h[o];
            for (int k = 0; k < 4; ++k) s += Wng[o * 4 + k] * gv[k];
            gb_n[g * 16 + o] = s;
        }
    }
}

// ---------------- K5: hidden edge block  e2 = relu([e1 | n1[senders]] @ B + gbias_e[graph]) ----------------
__global__ __launch_bounds__(256) void k_edge_hid(
    const _Float16* __restrict__ e1, const _Float16* __restrict__ n1,
    const int* __restrict__ senders,
    const float* __restrict__ gb_e, const _Float16* __restrict__ Bp,
    _Float16* __restrict__ e2)
{
    __shared__ __align__(16) _Float16 sA[128][32];
    __shared__ int sSend[128];
    const int tid = threadIdx.x;
    const long e0 = (long)blockIdx.x * 128;
    const int g = (int)(e0 / EPG);          // EPG % 128 == 0
    if (tid < 128) sSend[tid] = senders[e0 + tid];
    __syncthreads();
    if (tid < 128) *(v8h*)&sA[tid][0] = *(const v8h*)&e1[(e0 + tid) * 8];
    {
        int row = tid >> 1, half = tid & 1;
        *(v8h*)&sA[row][8 + 8 * half] = *(const v8h*)&n1[(long)sSend[row] * 16 + 8 * half];
    }
    if (tid < 128) { v8h z = {}; *(v8h*)&sA[tid][24] = z; }
    __syncthreads();
    const int wave = tid >> 5, lane = tid & 31;
    const int laneHi = lane >> 4, col = lane & 15;
    const _Float16* rowp = &sA[wave * 16 + col][0];
    const v16h* B = (const v16h*)Bp;
    v16h a = load_a_frag(rowp, 0, laneHi);
    v16h b = B[5 * 32 + lane];
    v8f c = {};
    c = __builtin_amdgcn_wmma_f32_16x16x32_f16(false, a, false, b, (short)0, c, false, false);
    const float bias = gb_e[g * 8 + (col & 7)];
    __syncthreads();
#pragma unroll
    for (int r = 0; r < 8; ++r)
        sA[wave * 16 + r + 8 * laneHi][col] = (_Float16)fmaxf(c[r] + bias, 0.f);
    __syncthreads();
    if (tid < 128) *(v8h*)&e2[(e0 + tid) * 8] = *(const v8h*)&sA[tid][0];
}

// ---------------- K7: hidden node block  n2 = relu([n1 | inc2] @ B + gbias_n[graph]) ----------------
__global__ __launch_bounds__(256) void k_node_hid(
    const _Float16* __restrict__ n1, const float* __restrict__ inc,
    const float* __restrict__ gb_n, const _Float16* __restrict__ Bp,
    _Float16* __restrict__ n2)
{
    __shared__ __align__(16) _Float16 sA[128][32];
    const int tid = threadIdx.x;
    const long n0 = (long)blockIdx.x * 128;
    const int g = (int)(n0 / NPG);          // NPG % 128 == 64: graph of the block's first row
    {
        int row = tid >> 1, half = tid & 1;
        *(v8h*)&sA[row][8 * half] = *(const v8h*)&n1[(n0 + row) * 16 + 8 * half];
    }
    for (int idx = tid; idx < 128 * 2; idx += 256) {
        int row = idx >> 1, q = idx & 1;
        const float4 f = *(const float4*)&inc[(n0 + row) * 8 + 4 * q];
        *(v4h*)&sA[row][16 + 4 * q] = cvt4(f);
    }
    if (tid < 128) { v8h z = {}; *(v8h*)&sA[tid][24] = z; }
    __syncthreads();
    const int wave = tid >> 5, lane = tid & 31;
    const int laneHi = lane >> 4, col = lane & 15;
    const _Float16* rowp = &sA[wave * 16 + col][0];
    const v16h* B = (const v16h*)Bp;
    v16h a = load_a_frag(rowp, 0, laneHi);
    v16h b = B[6 * 32 + lane];
    v8f c = {};
    c = __builtin_amdgcn_wmma_f32_16x16x32_f16(false, a, false, b, (short)0, c, false, false);
    const float bias = gb_n[g * 16 + col];
    __syncthreads();
#pragma unroll
    for (int r = 0; r < 8; ++r)
        sA[wave * 16 + r + 8 * laneHi][col] = (_Float16)fmaxf(c[r] + bias, 0.f);
    __syncthreads();
    {
        int row = tid >> 1, half = tid & 1;
        *(v8h*)&n2[(n0 + row) * 16 + 8 * half] = *(const v8h*)&sA[row][8 * half];
    }
}

// ---------------- K8: hidden global block + global readout ----------------
__global__ __launch_bounds__(256) void k_glob_hid(
    const _Float16* __restrict__ e2, const _Float16* __restrict__ n2,
    const float* __restrict__ g1,
    const float* __restrict__ Wge, const float* __restrict__ Wgn,
    const float* __restrict__ Wgg, const float* __restrict__ bg,
    const float* __restrict__ roWg, const float* __restrict__ robg,
    float* __restrict__ out)
{
    __shared__ float acc[24];
    const int g = blockIdx.x, tid = threadIdx.x;
    if (tid < 24) acc[tid] = 0.f;
    __syncthreads();
    float es[8]; for (int j = 0; j < 8; ++j) es[j] = 0.f;
    const long ebase = (long)g * EPG;
    for (int i = tid; i < EPG; i += 256) {
        const v8h ev = *(const v8h*)&e2[(ebase + i) * 8];
#pragma unroll
        for (int j = 0; j < 8; ++j) es[j] += (float)ev[j];
    }
    float ns[16]; for (int j = 0; j < 16; ++j) ns[j] = 0.f;
    const long nbase = (long)g * NPG;
    for (int i = tid; i < NPG; i += 256) {
        const v8h a = *(const v8h*)&n2[(nbase + i) * 16];
        const v8h b = *(const v8h*)&n2[(nbase + i) * 16 + 8];
#pragma unroll
        for (int j = 0; j < 8; ++j) { ns[j] += (float)a[j]; ns[8 + j] += (float)b[j]; }
    }
#pragma unroll
    for (int j = 0; j < 8; ++j) atomicAdd(&acc[j], es[j]);
#pragma unroll
    for (int j = 0; j < 16; ++j) atomicAdd(&acc[8 + j], ns[j]);
    __syncthreads();
    if (tid == 0) {
        float g2[4];
        for (int o = 0; o < 4; ++o) {
            float s = bg[o];
            for (int j = 0; j < 8;  ++j) s += Wge[o * 8 + j]  * (acc[j] / (float)EPG);
            for (int j = 0; j < 16; ++j) s += Wgn[o * 16 + j] * (acc[8 + j] / (float)NPG);
            for (int j = 0; j < 4;  ++j) s += Wgg[o * 4 + j]  * g1[g * 4 + j];
            g2[o] = fmaxf(s, 0.f);
        }
        float s = robg[0];
        for (int j = 0; j < 4; ++j) s += roWg[j] * g2[j];
        out[N_NODES + g] = sigmoidf_(s);
    }
}

// ---------------- K9: node readout ----------------
__global__ __launch_bounds__(256) void k_node_ro(
    const _Float16* __restrict__ n2, const float* __restrict__ W,
    const float* __restrict__ b, float* __restrict__ out)
{
    long i = (long)blockIdx.x * 256 + threadIdx.x;
    if (i < N_NODES) {
        const v8h a = *(const v8h*)&n2[i * 16];
        const v8h c = *(const v8h*)&n2[i * 16 + 8];
        float s = b[0];
#pragma unroll
        for (int j = 0; j < 8; ++j) s += W[j] * (float)a[j] + W[8 + j] * (float)c[j];
        out[i] = sigmoidf_(s);
    }
}

extern "C" void kernel_launch(void* const* d_in, const int* in_sizes, int n_in,
                              void* d_out, int out_size, void* d_ws, size_t ws_size,
                              hipStream_t stream)
{
    (void)in_sizes; (void)n_in; (void)out_size; (void)ws_size;
    const float* nf        = (const float*)d_in[0];
    const float* ef        = (const float*)d_in[1];
    const int*   senders   = (const int*)d_in[2];
    const int*   receivers = (const int*)d_in[3];
    const float* enc_Wee = (const float*)d_in[6];
    const float* enc_Wes = (const float*)d_in[7];
    const float* enc_be  = (const float*)d_in[8];
    const float* enc_Wnn = (const float*)d_in[9];
    const float* enc_Wni = (const float*)d_in[10];
    const float* enc_bn  = (const float*)d_in[11];
    const float* enc_Wge = (const float*)d_in[12];
    const float* enc_Wgn = (const float*)d_in[13];
    const float* enc_bg  = (const float*)d_in[14];
    const float* hid_Wee = (const float*)d_in[15];
    const float* hid_Wes = (const float*)d_in[16];
    const float* hid_Weg = (const float*)d_in[17];
    const float* hid_be  = (const float*)d_in[18];
    const float* hid_Wnn = (const float*)d_in[19];
    const float* hid_Wni = (const float*)d_in[20];
    const float* hid_Wng = (const float*)d_in[21];
    const float* hid_bn  = (const float*)d_in[22];
    const float* hid_Wge = (const float*)d_in[23];
    const float* hid_Wgn = (const float*)d_in[24];
    const float* hid_Wgg = (const float*)d_in[25];
    const float* hid_bg  = (const float*)d_in[26];
    const float* ro_Wn   = (const float*)d_in[27];
    const float* ro_bn   = (const float*)d_in[28];
    const float* ro_Wg   = (const float*)d_in[29];
    const float* ro_bg   = (const float*)d_in[30];
    float* out = (float*)d_out;

    char* ws = (char*)d_ws;
    size_t off = 0;
    _Float16* e1 = (_Float16*)(ws + off); off += (size_t)N_EDGES * 8 * 2;
    _Float16* e2 = (_Float16*)(ws + off); off += (size_t)N_EDGES * 8 * 2;
    float*    inc = (float*)(ws + off);   off += (size_t)N_NODES * 8 * 4;
    _Float16* n1 = (_Float16*)(ws + off); off += (size_t)N_NODES * 16 * 2;
    _Float16* n2 = (_Float16*)(ws + off); off += (size_t)N_NODES * 16 * 2;
    float*    g1 = (float*)(ws + off);    off += (size_t)N_GRAPH * 4 * 4;
    float*    gb_e = (float*)(ws + off);  off += (size_t)N_GRAPH * 8 * 4;
    float*    gb_n = (float*)(ws + off);  off += (size_t)N_GRAPH * 16 * 4;
    _Float16* Bp = (_Float16*)(ws + off); off += (size_t)7 * 32 * 16 * 2;

    k_pack_b<<<1, 256, 0, stream>>>(enc_Wee, enc_Wes, enc_Wnn, enc_Wni,
                                    hid_Wee, hid_Wes, hid_Wnn, hid_Wni, Bp);
    k_edge_enc<<<N_EDGES / 128, 256, 0, stream>>>(nf, ef, senders, enc_be, Bp, e1);
    k_inc_mean<<<N_GRAPH, 256, 0, stream>>>(e1, receivers, inc);
    k_node_enc<<<N_NODES / 128, 256, 0, stream>>>(nf, inc, enc_bn, Bp, n1);
    k_glob_enc<<<N_GRAPH, 256, 0, stream>>>(e1, n1, enc_Wge, enc_Wgn, enc_bg,
                                            hid_Weg, hid_be, hid_Wng, hid_bn, g1, gb_e, gb_n);
    k_edge_hid<<<N_EDGES / 128, 256, 0, stream>>>(e1, n1, senders, gb_e, Bp, e2);
    k_inc_mean<<<N_GRAPH, 256, 0, stream>>>(e2, receivers, inc);
    k_node_hid<<<N_NODES / 128, 256, 0, stream>>>(n1, inc, gb_n, Bp, n2);
    k_glob_hid<<<N_GRAPH, 256, 0, stream>>>(e2, n2, g1, hid_Wge, hid_Wgn, hid_Wgg, hid_bg,
                                            ro_Wg, ro_bg, out);
    k_node_ro<<<(N_NODES + 255) / 256, 256, 0, stream>>>(n2, ro_Wn, ro_bn, out);
}
